// EnBaseLayer_59184649339215
// MI455X (gfx1250) — matrix operand order repaired and checked by reading, not compiled, not run on hardware
//
#include <hip/hip_runtime.h>
#include <hip/hip_bf16.h>

// ---------------------------------------------------------------------------
// EGNN layer for MI455X (gfx1250): bf16 WMMA 16x16x32 for all GEMMs,
// f32 accumulate, atomic-f32 segment sums.
// Round 2: 64-edge M tiles (4x B-fragment reuse -> ~64 FLOP/byte from L2),
// LDS row padding for conflict-free ds_load_b128 fragments.
// ---------------------------------------------------------------------------

typedef __attribute__((ext_vector_type(16))) __bf16 v16bf;
typedef __attribute__((ext_vector_type(8)))  float  v8f;

constexpr int NND    = 20000;   // nodes   (20000/32 = 625 node tiles, exact)
constexpr int EED    = 320000;  // edges   (320000/64 = 5000 edge tiles, exact)
constexpr int HD     = 128;     // hidden dim
constexpr int KGAUSS = 20;
constexpr int DIN    = 304;     // 28 + 20 + 128 + 128
constexpr int DPAD   = 320;     // K padded to multiple of 32

// LDS row strides (bf16 elems): stride_dwords % 8 == 4 -> 16 lanes of a
// b128 A-fragment load start on distinct 4-dword bank spans (conflict-free),
// and stride bytes stay 16B-aligned.
constexpr int FSTR = 328;       // for 320-wide feature tile (656 B/row)
constexpr int HSTR = 136;       // for 128-wide tiles        (272 B/row)
constexpr int NSTR = 264;       // for 256-wide node tile    (528 B/row)

constexpr int EM = 64;          // edges per block (4 M subtiles)
constexpr int NM = 32;          // nodes per block (2 M subtiles)

// ---- workspace layout (bytes) ----
constexpr size_t OFF_MI   = 0;
constexpr size_t SZ_MI    = (size_t)NND * HD * 4;          // mi accumulator
constexpr size_t OFF_XA   = OFF_MI + SZ_MI;                // x accumulator
constexpr size_t SZ_XA    = (size_t)NND * 3 * 4;
constexpr size_t OFF_EW1T = (OFF_XA + SZ_XA + 255) & ~(size_t)255;
constexpr size_t SZ_EW1T  = (size_t)HD * DPAD * 2;         // bf16 [128][320]
constexpr size_t OFF_EW2T = OFF_EW1T + SZ_EW1T;
constexpr size_t SZ_HH    = (size_t)HD * HD * 2;           // bf16 [128][128]
constexpr size_t OFF_NW1T = OFF_EW2T + SZ_HH;
constexpr size_t SZ_NW1T  = (size_t)HD * 2 * HD * 2;       // bf16 [128][256]
constexpr size_t OFF_NW2T = OFF_NW1T + SZ_NW1T;
constexpr size_t OFF_XW1T = OFF_NW2T + SZ_HH;

// round-to-nearest-even f32 -> bf16 (pure integer)
__device__ __forceinline__ __bf16 f2bf(float f) {
  unsigned u = __builtin_bit_cast(unsigned, f);
  u += 0x7FFFu + ((u >> 16) & 1u);
  unsigned short s = (unsigned short)(u >> 16);
  return __builtin_bit_cast(__bf16, s);
}
__device__ __forceinline__ float bf2f(__bf16 b) {
  unsigned u = (unsigned)__builtin_bit_cast(unsigned short, b) << 16;
  return __builtin_bit_cast(float, u);
}

// A fragment (16x32 bf16). Per lane: row m = lane&15, half = lane>>4.
// Element i -> K = (i<8 ? i : i+8) + half*8 relative to chunk base.
// p points at &row[kchunk + half*8]; two 16B loads at +0 and +32 bytes.
__device__ __forceinline__ v16bf load_a_frag(const __bf16* p) {
  union { uint4 u[2]; v16bf v; } c;
  c.u[0] = *(const uint4*)(p);
  c.u[1] = *(const uint4*)(p + 16);
  return c.v;
}

// B fragment (32x16 bf16, K-major column per lane). Per lane: col n = lane&15,
// half = lane>>4; element i -> K = (i<8 ? i : i+8) + half*8. p = &col[kchunk + half*8];
// two 16B loads at +0 and +32 bytes.
__device__ __forceinline__ v16bf load_b_frag(const __bf16* p) {
  union { uint4 u[2]; v16bf v; } c;
  c.u[0] = *(const uint4*)(p);
  c.u[1] = *(const uint4*)(p + 16);
  return c.v;
}

// ---------------------------------------------------------------------------
__global__ void zero_kernel(float* __restrict__ p, int n) {
  int t = blockIdx.x * blockDim.x + threadIdx.x;
  if (t < n) p[t] = 0.0f;
}

// Transpose + bf16-convert all weights: wT[n*K + k] = w[k*N + n]
__global__ void convert_weights(
    const float* __restrict__ ew1, const float* __restrict__ ew2,
    const float* __restrict__ nw1, const float* __restrict__ nw2,
    const float* __restrict__ xw1,
    __bf16* __restrict__ ew1T, __bf16* __restrict__ ew2T,
    __bf16* __restrict__ nw1T, __bf16* __restrict__ nw2T,
    __bf16* __restrict__ xw1T) {
  int t = blockIdx.x * blockDim.x + threadIdx.x;
  if (t < HD * DPAD) {                       // ew1: [304x128] -> [128][320] pad
    int n = t / DPAD, k = t % DPAD;
    ew1T[t] = (k < DIN) ? f2bf(ew1[k * HD + n]) : f2bf(0.0f);
    return;
  }
  t -= HD * DPAD;
  if (t < HD * HD) { int n = t / HD, k = t % HD; ew2T[t] = f2bf(ew2[k * HD + n]); return; }
  t -= HD * HD;
  if (t < HD * 2 * HD) { int n = t / (2 * HD), k = t % (2 * HD); nw1T[t] = f2bf(nw1[k * HD + n]); return; }
  t -= HD * 2 * HD;
  if (t < HD * HD) { int n = t / HD, k = t % HD; nw2T[t] = f2bf(nw2[k * HD + n]); return; }
  t -= HD * HD;
  if (t < HD * HD) { int n = t / HD, k = t % HD; xw1T[t] = f2bf(xw1[k * HD + n]); return; }
}

// ---------------------------------------------------------------------------
// Edge kernel: one block = 64 edges (4 M subtiles); 8 waves each own a
// 16-wide N tile and reuse every B fragment across the 4 M subtiles.
__global__ __launch_bounds__(256) void edge_kernel(
    const float* __restrict__ hptr, const float* __restrict__ xptr,
    const float* __restrict__ edge_attr,
    const __bf16* __restrict__ ew1T, const float* __restrict__ eb1,
    const __bf16* __restrict__ ew2T, const float* __restrict__ eb2,
    const float* __restrict__ inf_w, const float* __restrict__ inf_b,
    const __bf16* __restrict__ xw1T, const float* __restrict__ xb1,
    const float* __restrict__ xw2, const float* __restrict__ xb2,
    const int* __restrict__ edge_index,
    float* __restrict__ miacc, float* __restrict__ xacc) {
  __shared__ alignas(16) __bf16 featb[EM][FSTR];   // 64 x 328 bf16
  __shared__ alignas(16) __bf16 midb[EM][HSTR];    // relu(feat@ew1+b)
  __shared__ alignas(16) __bf16 mijb[EM][HSTR];    // relu(mid@ew2+b)
  __shared__ alignas(16) __bf16 t2b[EM][HSTR];     // relu(mij@xw1+b)
  __shared__ float relx[EM][3];
  __shared__ int   dstl[EM];
  __shared__ float scl[EM];
  __shared__ float xgs[EM];

  const int t  = threadIdx.x;
  const int e0 = blockIdx.x * EM;

  // ---- phase 0: build bf16 feature tile [64][320] in LDS
  if (t < EM) {
    const int e = e0 + t;
    const int d = edge_index[e];
    const int s = edge_index[EED + e];
    dstl[t] = d;
    const float rx = xptr[d * 3 + 0] - xptr[s * 3 + 0];
    const float ry = xptr[d * 3 + 1] - xptr[s * 3 + 1];
    const float rz = xptr[d * 3 + 2] - xptr[s * 3 + 2];
    relx[t][0] = rx; relx[t][1] = ry; relx[t][2] = rz;
    const float l2 = sqrtf(rx * rx + ry * ry + rz * rz);
    const float step  = 100.0f / 19.0f;          // linspace(0,100,20) spacing
    const float coeff = -0.5f / (step * step);
    #pragma unroll
    for (int k = 0; k < KGAUSS; ++k) {
      const float dlt = l2 - step * (float)k;
      featb[t][28 + k] = f2bf(__expf(coeff * dlt * dlt));
    }
    #pragma unroll
    for (int k = 0; k < 28; ++k) featb[t][k] = f2bf(edge_attr[e * 28 + k]);
    #pragma unroll
    for (int k = DIN; k < DPAD; ++k) featb[t][k] = f2bf(0.0f);
  }
  for (int idx = t; idx < EM * HD; idx += 256) {  // gather h[dst], h[src]
    const int row = idx >> 7, col = idx & 127;
    const int e = e0 + row;
    const int d = edge_index[e];
    const int s = edge_index[EED + e];
    featb[row][48 + col]  = f2bf(hptr[d * HD + col]);
    featb[row][176 + col] = f2bf(hptr[s * HD + col]);
  }
  __syncthreads();

  const int lane = t & 31;
  const int hl   = lane >> 4;       // K-half of this lane
  const int nloc = lane & 15;       // output column within tile
  const int am   = lane & 15;       // A row within M subtile
  const int n0   = (t >> 5) * 16;   // this wave's N tile origin

  // ---- layer 1: mid = relu(feat @ ew1 + eb1), K=320, 4 M subtiles
  {
    v8f acc0 = {}, acc1 = {}, acc2 = {}, acc3 = {};
    #pragma unroll
    for (int kc = 0; kc < DPAD / 32; ++kc) {
      v16bf b = load_b_frag(&ew1T[(n0 + nloc) * DPAD + kc * 32 + hl * 8]);
      v16bf a0 = load_a_frag(&featb[ 0 + am][kc * 32 + hl * 8]);
      v16bf a1 = load_a_frag(&featb[16 + am][kc * 32 + hl * 8]);
      v16bf a2 = load_a_frag(&featb[32 + am][kc * 32 + hl * 8]);
      v16bf a3 = load_a_frag(&featb[48 + am][kc * 32 + hl * 8]);
      acc0 = __builtin_amdgcn_wmma_f32_16x16x32_bf16(false, a0, false, b, (short)0, acc0, false, false);
      acc1 = __builtin_amdgcn_wmma_f32_16x16x32_bf16(false, a1, false, b, (short)0, acc1, false, false);
      acc2 = __builtin_amdgcn_wmma_f32_16x16x32_bf16(false, a2, false, b, (short)0, acc2, false, false);
      acc3 = __builtin_amdgcn_wmma_f32_16x16x32_bf16(false, a3, false, b, (short)0, acc3, false, false);
    }
    const float bias = eb1[n0 + nloc];
    #pragma unroll
    for (int r = 0; r < 8; ++r) {
      const int m = r + hl * 8;
      midb[ 0 + m][n0 + nloc] = f2bf(fmaxf(acc0[r] + bias, 0.0f));
      midb[16 + m][n0 + nloc] = f2bf(fmaxf(acc1[r] + bias, 0.0f));
      midb[32 + m][n0 + nloc] = f2bf(fmaxf(acc2[r] + bias, 0.0f));
      midb[48 + m][n0 + nloc] = f2bf(fmaxf(acc3[r] + bias, 0.0f));
    }
  }
  __syncthreads();

  // ---- layer 2: mij = relu(mid @ ew2 + eb2), K=128
  {
    v8f acc0 = {}, acc1 = {}, acc2 = {}, acc3 = {};
    #pragma unroll
    for (int kc = 0; kc < HD / 32; ++kc) {
      v16bf b = load_b_frag(&ew2T[(n0 + nloc) * HD + kc * 32 + hl * 8]);
      v16bf a0 = load_a_frag(&midb[ 0 + am][kc * 32 + hl * 8]);
      v16bf a1 = load_a_frag(&midb[16 + am][kc * 32 + hl * 8]);
      v16bf a2 = load_a_frag(&midb[32 + am][kc * 32 + hl * 8]);
      v16bf a3 = load_a_frag(&midb[48 + am][kc * 32 + hl * 8]);
      acc0 = __builtin_amdgcn_wmma_f32_16x16x32_bf16(false, a0, false, b, (short)0, acc0, false, false);
      acc1 = __builtin_amdgcn_wmma_f32_16x16x32_bf16(false, a1, false, b, (short)0, acc1, false, false);
      acc2 = __builtin_amdgcn_wmma_f32_16x16x32_bf16(false, a2, false, b, (short)0, acc2, false, false);
      acc3 = __builtin_amdgcn_wmma_f32_16x16x32_bf16(false, a3, false, b, (short)0, acc3, false, false);
    }
    const float bias = eb2[n0 + nloc];
    #pragma unroll
    for (int r = 0; r < 8; ++r) {
      const int m = r + hl * 8;
      mijb[ 0 + m][n0 + nloc] = f2bf(fmaxf(acc0[r] + bias, 0.0f));
      mijb[16 + m][n0 + nloc] = f2bf(fmaxf(acc1[r] + bias, 0.0f));
      mijb[32 + m][n0 + nloc] = f2bf(fmaxf(acc2[r] + bias, 0.0f));
      mijb[48 + m][n0 + nloc] = f2bf(fmaxf(acc3[r] + bias, 0.0f));
    }
  }
  __syncthreads();

  // ---- x path hidden: t2 = relu(mij @ xw1 + xb1), K=128
  {
    v8f acc0 = {}, acc1 = {}, acc2 = {}, acc3 = {};
    #pragma unroll
    for (int kc = 0; kc < HD / 32; ++kc) {
      v16bf b = load_b_frag(&xw1T[(n0 + nloc) * HD + kc * 32 + hl * 8]);
      v16bf a0 = load_a_frag(&mijb[ 0 + am][kc * 32 + hl * 8]);
      v16bf a1 = load_a_frag(&mijb[16 + am][kc * 32 + hl * 8]);
      v16bf a2 = load_a_frag(&mijb[32 + am][kc * 32 + hl * 8]);
      v16bf a3 = load_a_frag(&mijb[48 + am][kc * 32 + hl * 8]);
      acc0 = __builtin_amdgcn_wmma_f32_16x16x32_bf16(false, a0, false, b, (short)0, acc0, false, false);
      acc1 = __builtin_amdgcn_wmma_f32_16x16x32_bf16(false, a1, false, b, (short)0, acc1, false, false);
      acc2 = __builtin_amdgcn_wmma_f32_16x16x32_bf16(false, a2, false, b, (short)0, acc2, false, false);
      acc3 = __builtin_amdgcn_wmma_f32_16x16x32_bf16(false, a3, false, b, (short)0, acc3, false, false);
    }
    const float bias = xb1[n0 + nloc];
    #pragma unroll
    for (int r = 0; r < 8; ++r) {
      const int m = r + hl * 8;
      t2b[ 0 + m][n0 + nloc] = f2bf(fmaxf(acc0[r] + bias, 0.0f));
      t2b[16 + m][n0 + nloc] = f2bf(fmaxf(acc1[r] + bias, 0.0f));
      t2b[32 + m][n0 + nloc] = f2bf(fmaxf(acc2[r] + bias, 0.0f));
      t2b[48 + m][n0 + nloc] = f2bf(fmaxf(acc3[r] + bias, 0.0f));
    }
  }
  __syncthreads();

  // ---- H->1 projections: eij (sigmoid) and xg, one lane per edge
  if (t < EM) {
    float s1 = inf_b[0];
    float s2 = xb2[0];
    for (int k = 0; k < HD; ++k) {
      s1 += bf2f(mijb[t][k]) * inf_w[k];
      s2 += bf2f(t2b[t][k]) * xw2[k];
    }
    scl[t] = 1.0f / (1.0f + __expf(-s1));
    xgs[t] = s2;
  }
  __syncthreads();

  // ---- scatter: segment sums via f32 atomics
  for (int idx = t; idx < EM * HD; idx += 256) {
    const int row = idx >> 7, col = idx & 127;
    atomicAdd(&miacc[dstl[row] * HD + col], bf2f(mijb[row][col]) * scl[row]);
  }
  if (t < EM * 3) {
    const int row = t / 3, c = t % 3;
    atomicAdd(&xacc[dstl[row] * 3 + c], relx[row][c] * xgs[row]);
  }
}

// ---------------------------------------------------------------------------
// Node kernel: out = relu([mi, h] @ nw1 + nb1) @ nw2 + nb2, 32 nodes/block.
__global__ __launch_bounds__(256) void node_kernel(
    const float* __restrict__ hptr, const float* __restrict__ miacc,
    const __bf16* __restrict__ nw1T, const float* __restrict__ nb1,
    const __bf16* __restrict__ nw2T, const float* __restrict__ nb2,
    float* __restrict__ out) {
  __shared__ alignas(16) __bf16 featb[NM][NSTR];   // 32 x 264 bf16
  __shared__ alignas(16) __bf16 midb[NM][HSTR];

  const int t  = threadIdx.x;
  const int v0 = blockIdx.x * NM;

  for (int idx = t; idx < NM * HD; idx += 256) {
    const int row = idx >> 7, col = idx & 127;
    const int node = v0 + row;
    featb[row][col]      = f2bf(miacc[node * HD + col]);  // concat: mi first
    featb[row][HD + col] = f2bf(hptr[node * HD + col]);   // then h
  }
  __syncthreads();

  const int lane = t & 31;
  const int hl   = lane >> 4;
  const int nloc = lane & 15;
  const int am   = lane & 15;
  const int n0   = (t >> 5) * 16;

  {
    v8f acc0 = {}, acc1 = {};
    #pragma unroll
    for (int kc = 0; kc < (2 * HD) / 32; ++kc) {
      v16bf b = load_b_frag(&nw1T[(n0 + nloc) * 2 * HD + kc * 32 + hl * 8]);
      v16bf a0 = load_a_frag(&featb[ 0 + am][kc * 32 + hl * 8]);
      v16bf a1 = load_a_frag(&featb[16 + am][kc * 32 + hl * 8]);
      acc0 = __builtin_amdgcn_wmma_f32_16x16x32_bf16(false, a0, false, b, (short)0, acc0, false, false);
      acc1 = __builtin_amdgcn_wmma_f32_16x16x32_bf16(false, a1, false, b, (short)0, acc1, false, false);
    }
    const float bias = nb1[n0 + nloc];
    #pragma unroll
    for (int r = 0; r < 8; ++r) {
      const int m = r + hl * 8;
      midb[ 0 + m][n0 + nloc] = f2bf(fmaxf(acc0[r] + bias, 0.0f));
      midb[16 + m][n0 + nloc] = f2bf(fmaxf(acc1[r] + bias, 0.0f));
    }
  }
  __syncthreads();

  {
    v8f acc0 = {}, acc1 = {};
    #pragma unroll
    for (int kc = 0; kc < HD / 32; ++kc) {
      v16bf b = load_b_frag(&nw2T[(n0 + nloc) * HD + kc * 32 + hl * 8]);
      v16bf a0 = load_a_frag(&midb[ 0 + am][kc * 32 + hl * 8]);
      v16bf a1 = load_a_frag(&midb[16 + am][kc * 32 + hl * 8]);
      acc0 = __builtin_amdgcn_wmma_f32_16x16x32_bf16(false, a0, false, b, (short)0, acc0, false, false);
      acc1 = __builtin_amdgcn_wmma_f32_16x16x32_bf16(false, a1, false, b, (short)0, acc1, false, false);
    }
    const float bias = nb2[n0 + nloc];  // no final relu
    #pragma unroll
    for (int r = 0; r < 8; ++r) {
      const int m = r + hl * 8;
      out[(v0 +  0 + m) * HD + n0 + nloc] = acc0[r] + bias;
      out[(v0 + 16 + m) * HD + n0 + nloc] = acc1[r] + bias;
    }
  }
}

// ---------------------------------------------------------------------------
__global__ void xout_kernel(const float* __restrict__ x,
                            const float* __restrict__ xacc,
                            float* __restrict__ xout) {
  int t = blockIdx.x * blockDim.x + threadIdx.x;
  if (t < NND * 3) xout[t] = x[t] + xacc[t] * (1.0f / (float)EED);
}

// ---------------------------------------------------------------------------
extern "C" void kernel_launch(void* const* d_in, const int* in_sizes, int n_in,
                              void* d_out, int out_size, void* d_ws, size_t ws_size,
                              hipStream_t stream) {
  const float* h         = (const float*)d_in[0];
  const float* x         = (const float*)d_in[1];
  const float* edge_attr = (const float*)d_in[2];
  const float* ew1   = (const float*)d_in[3];
  const float* eb1   = (const float*)d_in[4];
  const float* ew2   = (const float*)d_in[5];
  const float* eb2   = (const float*)d_in[6];
  const float* inf_w = (const float*)d_in[7];
  const float* inf_b = (const float*)d_in[8];
  const float* nw1   = (const float*)d_in[9];
  const float* nb1   = (const float*)d_in[10];
  const float* nw2   = (const float*)d_in[11];
  const float* nb2   = (const float*)d_in[12];
  const float* xw1   = (const float*)d_in[13];
  const float* xb1   = (const float*)d_in[14];
  const float* xw2   = (const float*)d_in[15];
  const float* xb2   = (const float*)d_in[16];
  const int* edge_index = (const int*)d_in[17];
  float* out = (float*)d_out;

  char* ws = (char*)d_ws;
  float*  miacc = (float*)(ws + OFF_MI);
  float*  xacc  = (float*)(ws + OFF_XA);
  __bf16* ew1T  = (__bf16*)(ws + OFF_EW1T);
  __bf16* ew2T  = (__bf16*)(ws + OFF_EW2T);
  __bf16* nw1T  = (__bf16*)(ws + OFF_NW1T);
  __bf16* nw2T  = (__bf16*)(ws + OFF_NW2T);
  __bf16* xw1T  = (__bf16*)(ws + OFF_XW1T);

  // 1) zero accumulators (mi + xacc are contiguous)
  {
    const int n = NND * HD + NND * 3;
    zero_kernel<<<(n + 255) / 256, 256, 0, stream>>>(miacc, n);
  }
  // 2) transpose+convert weights to bf16
  {
    const int n = HD * DPAD + HD * HD + HD * 2 * HD + HD * HD + HD * HD;
    convert_weights<<<(n + 255) / 256, 256, 0, stream>>>(
        ew1, ew2, nw1, nw2, xw1, ew1T, ew2T, nw1T, nw2T, xw1T);
  }
  // 3) edge MLP + scatter (E/64 = 5000 tiles)
  edge_kernel<<<EED / EM, 256, 0, stream>>>(
      h, x, edge_attr, ew1T, eb1, ew2T, eb2, inf_w, inf_b,
      xw1T, xb1, xw2, xb2, edge_index, miacc, xacc);
  // 4) node MLP (N/32 = 625 tiles)
  node_kernel<<<NND / NM, 256, 0, stream>>>(h, miacc, nw1T, nb1, nw2T, nb2, out);
  // 5) coordinate update into second output slot
  xout_kernel<<<(NND * 3 + 255) / 256, 256, 0, stream>>>(x, xacc, out + (size_t)NND * HD);
}
